// NeighborhoodAttention1D_59390807769419
// MI455X (gfx1250) — matrix operand order repaired and checked by reading, not compiled, not run on hardware
//
#include <hip/hip_runtime.h>
#include <hip/hip_bf16.h>

typedef __attribute__((ext_vector_type(16))) __bf16 v16bf;
typedef __attribute__((ext_vector_type(8)))  __bf16 v8bf;
typedef __attribute__((ext_vector_type(8)))  float  v8f;
typedef __attribute__((ext_vector_type(4)))  int    v4i;

#define NUM_HEADS 16
#define KSZ 13
#define LSEQ 4096
#define CDIM 512
#define BDIM 4
#define DH 32
#define MROWS (BDIM * LSEQ)   // 16384

#if defined(__has_builtin)
#if __has_builtin(__builtin_amdgcn_global_load_async_to_lds_b128)
#define HAVE_ASYNC_LDS 1
#endif
#endif
#ifndef HAVE_ASYNC_LDS
#define HAVE_ASYNC_LDS 0
#endif

#if HAVE_ASYNC_LDS
typedef __attribute__((address_space(1))) v4i as1_v4i;
typedef __attribute__((address_space(3))) v4i as3_v4i;
__device__ __forceinline__ void cp_async16(const void* g, void* l) {
  __builtin_amdgcn_global_load_async_to_lds_b128((as1_v4i*)(void*)g,
                                                 (as3_v4i*)l, 0, 0);
}
#if __has_builtin(__builtin_amdgcn_s_wait_asynccnt)
#define WAIT_ASYNC(n) __builtin_amdgcn_s_wait_asynccnt(n)
#else
#define WAIT_ASYNC(n) asm volatile("s_wait_asynccnt %0" ::"i"(n) : "memory")
#endif
#endif

// ---------------- f32 -> bf16 conversion ----------------
__global__ __launch_bounds__(256) void cvt_f32_bf16(const float* __restrict__ in,
                                                    __bf16* __restrict__ out, int n) {
  int i = blockIdx.x * 256 + threadIdx.x;
  if (i < n) out[i] = (__bf16)in[i];
}

// ---------------- fragment builders (ISA 7.12.2 layouts) ----------------
__device__ __forceinline__ v16bf frag_from_lds(const __bf16* p, int off0, int off1) {
  v8bf lo = *(const v8bf*)(p + off0);
  v8bf hi = *(const v8bf*)(p + off1);
  return __builtin_shufflevector(lo, hi, 0, 1, 2, 3, 4, 5, 6, 7,
                                 8, 9, 10, 11, 12, 13, 14, 15);
}

// ---------------- tiled bf16 WMMA GEMM ----------------
// C[m,n] = sum_k A[m,k] * W[n,k] + bias[n];  M tiles of 256, N tiles of 64, BK=32.
// Block: 256 threads (8 waves); wave computes 32x64 (2 A-frags x 4 B-frags).
// MODE 0: scatter into q/k/v [B*H][L][32] f32 (q pre-scaled by Dh^-0.5)
// MODE 1: write f32 out[m*512 + n]
template <int MODE>
__global__ __launch_bounds__(256) void gemm_bf16_wmma(
    const __bf16* __restrict__ A,      // [M][512] row-major
    const __bf16* __restrict__ W,      // [N][512] row-major
    const float*  __restrict__ bias,   // [N]
    float* __restrict__ oq, float* __restrict__ ok, float* __restrict__ ov,
    float* __restrict__ oflat) {
  constexpr int BK = 32;
  constexpr int KD = CDIM;
  constexpr int NK = KD / BK;  // 16
  __shared__ __bf16 As[2][256][40];  // row stride 80B (16B aligned)
  __shared__ __bf16 Bs[2][64][40];

  const int tid  = threadIdx.x;
  const int wave = tid >> 5;
  const int lane = tid & 31;
  const int m0 = blockIdx.x * 256;
  const int n0 = blockIdx.y * 64;

  const __bf16* gA = A + (size_t)(m0 + tid) * KD;              // thread's A row
  const __bf16* gB = W + (size_t)(n0 + (tid >> 2)) * KD + (tid & 3) * 8;

  v8f zero = {0.f, 0.f, 0.f, 0.f, 0.f, 0.f, 0.f, 0.f};
  v8f acc[2][4] = {{zero, zero, zero, zero}, {zero, zero, zero, zero}};

  const int arow = wave * 32 + (lane & 15);
  const int akb  = (lane >> 4) << 3;   // A: K = 8*(lane>>4) + {0..7,16..23}
  const int bkb  = (lane >> 4) << 3;   // B: K = 8*(lane>>4) + {0..7,16..23}

#if HAVE_ASYNC_LDS
  // ---- async global->LDS pipeline (5 copies per thread per tile) ----
  {
    // tile 0 -> buffer 0
#pragma unroll
    for (int c = 0; c < 32; c += 8) cp_async16(gA + c, &As[0][tid][c]);
    cp_async16(gB, &Bs[0][tid >> 2][(tid & 3) * 8]);
  }
  for (int k = 0; k < NK; ++k) {
    if (k + 1 < NK) {
      const int k0 = (k + 1) * BK;
      const int nb = (k + 1) & 1;
#pragma unroll
      for (int c = 0; c < 32; c += 8) cp_async16(gA + k0 + c, &As[nb][tid][c]);
      cp_async16(gB + k0, &Bs[nb][tid >> 2][(tid & 3) * 8]);
      WAIT_ASYNC(5);   // in-order: tile k's 5 copies complete
    } else {
      WAIT_ASYNC(0);
    }
    __syncthreads();
    const int cb = k & 1;
    // load all fragments up front (distinct regs) so ds_loads issue as a clause
    v16bf a0 = frag_from_lds(&As[cb][arow][0],      akb, akb + 16);
    v16bf a1 = frag_from_lds(&As[cb][arow + 16][0], akb, akb + 16);
    v16bf bfr[4];
#pragma unroll
    for (int f = 0; f < 4; ++f)
      bfr[f] = frag_from_lds(&Bs[cb][f * 16 + (lane & 15)][0], bkb, bkb + 16);
#pragma unroll
    for (int f = 0; f < 4; ++f) {
      acc[0][f] = __builtin_amdgcn_wmma_f32_16x16x32_bf16(false, a0, false, bfr[f],
                                                          (short)0, acc[0][f], false, false);
      acc[1][f] = __builtin_amdgcn_wmma_f32_16x16x32_bf16(false, a1, false, bfr[f],
                                                          (short)0, acc[1][f], false, false);
    }
    __syncthreads();
  }
#else
  // ---- register-staged pipeline (distinct regs) ----
  uint4 ra[4], rb;
  auto loadregs = [&](int k0) {
#pragma unroll
    for (int i = 0; i < 4; ++i) ra[i] = *(const uint4*)(gA + k0 + i * 8);
    rb = *(const uint4*)(gB + k0);
  };
  auto storeregs = [&](int b) {
#pragma unroll
    for (int i = 0; i < 4; ++i) *(uint4*)&As[b][tid][i * 8] = ra[i];
    *(uint4*)&Bs[b][tid >> 2][(tid & 3) * 8] = rb;
  };
  loadregs(0);
  storeregs(0);
  if (NK > 1) loadregs(BK);
  for (int k = 0; k < NK; ++k) {
    __syncthreads();
    const int cb = k & 1;
    v16bf a0 = frag_from_lds(&As[cb][arow][0],      akb, akb + 16);
    v16bf a1 = frag_from_lds(&As[cb][arow + 16][0], akb, akb + 16);
    v16bf bfr[4];
#pragma unroll
    for (int f = 0; f < 4; ++f)
      bfr[f] = frag_from_lds(&Bs[cb][f * 16 + (lane & 15)][0], bkb, bkb + 16);
#pragma unroll
    for (int f = 0; f < 4; ++f) {
      acc[0][f] = __builtin_amdgcn_wmma_f32_16x16x32_bf16(false, a0, false, bfr[f],
                                                          (short)0, acc[0][f], false, false);
      acc[1][f] = __builtin_amdgcn_wmma_f32_16x16x32_bf16(false, a1, false, bfr[f],
                                                          (short)0, acc[1][f], false, false);
    }
    if (k + 1 < NK) {
      storeregs((k + 1) & 1);
      if (k + 2 < NK) loadregs((k + 2) * BK);
    }
  }
#endif

  // Epilogue. C/D layout: VGPR g -> row g + 8*(lane>>4); col = lane&15.
  const int ncl = lane & 15;
#pragma unroll
  for (int sub = 0; sub < 2; ++sub) {
    const int mrow_base = m0 + wave * 32 + sub * 16 + ((lane >> 4) << 3);
#pragma unroll
    for (int f = 0; f < 4; ++f) {
      const int ncol = n0 + f * 16 + ncl;
      const float bv = bias[ncol];
#pragma unroll
      for (int g = 0; g < 8; ++g) {
        const int mrow = mrow_base + g;
        float val = acc[sub][f][g] + bv;
        if (MODE == 0) {
          const int which = ncol >> 9;          // 0=q 1=k 2=v
          const int h = (ncol >> 5) & 15;
          const int d = ncol & 31;
          const int bb = mrow >> 12;            // m = b*L + l
          const int l = mrow & (LSEQ - 1);
          const size_t off = (((size_t)(bb * NUM_HEADS + h)) * LSEQ + l) * DH + d;
          if (which == 0)      oq[off] = val * 0.17677669529663689f;  // Dh^-0.5
          else if (which == 1) ok[off] = val;
          else                 ov[off] = val;
        } else {
          oflat[(size_t)mrow * CDIM + ncol] = val;
        }
      }
    }
  }
}

// ---------------- neighborhood attention (wave per query, lane = d) ----------------
__device__ __forceinline__ float wave_sum32(float v) {
#pragma unroll
  for (int o = 16; o > 0; o >>= 1) v += __shfl_xor(v, o, 32);
  return v;
}
__device__ __forceinline__ float wave_max32(float v) {
#pragma unroll
  for (int o = 16; o > 0; o >>= 1) v = fmaxf(v, __shfl_xor(v, o, 32));
  return v;
}

__global__ __launch_bounds__(256) void natten1d_kernel(
    const float* __restrict__ qT, const float* __restrict__ kT,
    const float* __restrict__ vT, const float* __restrict__ rpb,
    __bf16* __restrict__ yb) {
  const int wave = threadIdx.x >> 5;
  const int lane = threadIdx.x & 31;
  const int gq = blockIdx.x * 8 + wave;          // [0, B*H*L)
  const int bh = gq >> 12;                       // b*16 + h  (L = 4096)
  const int l  = gq & (LSEQ - 1);
  const int b  = bh >> 4;
  const int h  = bh & 15;

  int start = l - (KSZ / 2);
  start = start < 0 ? 0 : start;
  start = start > (LSEQ - KSZ) ? (LSEQ - KSZ) : start;

  const size_t rowbase = ((size_t)bh * LSEQ + start) * DH;
  const float q = qT[((size_t)bh * LSEQ + l) * DH + lane];

  float s = -3.0e38f;
#pragma unroll
  for (int j = 0; j < KSZ; ++j) {
    float t = q * kT[rowbase + (size_t)j * DH + lane];
    t = wave_sum32(t);
    if (lane == j) s = t;
  }
  if (lane < KSZ) s += rpb[h * (2 * KSZ - 1) + (start + lane - l + (KSZ - 1))];

  const float m = wave_max32(s);
  const float e = (lane < KSZ) ? __expf(s - m) : 0.f;
  const float denom = wave_sum32(e);
  const float p = e / denom;

  float o = 0.f;
#pragma unroll
  for (int j = 0; j < KSZ; ++j) {
    const float pj = __shfl(p, j, 32);
    o += pj * vT[rowbase + (size_t)j * DH + lane];
  }
  yb[((size_t)b * LSEQ + l) * CDIM + h * DH + lane] = (__bf16)o;
}

// ---------------- launcher ----------------
extern "C" void kernel_launch(void* const* d_in, const int* in_sizes, int n_in,
                              void* d_out, int out_size, void* d_ws, size_t ws_size,
                              hipStream_t stream) {
  const float* x      = (const float*)d_in[0];
  const float* qkv_w  = (const float*)d_in[1];
  const float* qkv_b  = (const float*)d_in[2];
  const float* rpb    = (const float*)d_in[3];
  const float* proj_w = (const float*)d_in[4];
  const float* proj_b = (const float*)d_in[5];
  float* out = (float*)d_out;

  const size_t nX  = (size_t)MROWS * CDIM;      // 8388608
  const size_t nWq = (size_t)3 * CDIM * CDIM;   // 786432
  const size_t nWp = (size_t)CDIM * CDIM;       // 262144

  char* ws = (char*)d_ws;
  size_t off = 0;
  __bf16* xb  = (__bf16*)(ws + off); off += nX * 2;
  __bf16* wqb = (__bf16*)(ws + off); off += nWq * 2;
  __bf16* wpb = (__bf16*)(ws + off); off += nWp * 2;
  float*  qT  = (float*)(ws + off);  off += nX * 4;
  float*  kT  = (float*)(ws + off);  off += nX * 4;
  float*  vT  = (float*)(ws + off);  off += nX * 4;
  __bf16* yb  = (__bf16*)(ws + off); off += nX * 2;

  cvt_f32_bf16<<<(nX + 255) / 256, 256, 0, stream>>>(x, xb, (int)nX);
  cvt_f32_bf16<<<(nWq + 255) / 256, 256, 0, stream>>>(qkv_w, wqb, (int)nWq);
  cvt_f32_bf16<<<(nWp + 255) / 256, 256, 0, stream>>>(proj_w, wpb, (int)nWp);

  // QKV GEMM: M=16384, N=1536
  gemm_bf16_wmma<0><<<dim3(MROWS / 256, (3 * CDIM) / 64), 256, 0, stream>>>(
      xb, wqb, qkv_b, qT, kT, vT, nullptr);

  // attention: B*H*L / 8 waves-per-block
  natten1d_kernel<<<(BDIM * NUM_HEADS * LSEQ) / 8, 256, 0, stream>>>(
      qT, kT, vT, rpb, yb);

  // Proj GEMM: M=16384, N=512 -> d_out
  gemm_bf16_wmma<1><<<dim3(MROWS / 256, CDIM / 64), 256, 0, stream>>>(
      yb, wpb, proj_b, nullptr, nullptr, nullptr, out);
}
